// Llama4MoE_33913061769720
// MI455X (gfx1250) — matrix operand order repaired and checked by reading, not compiled, not run on hardware
//
#include <hip/hip_runtime.h>
#include <hip/hip_bf16.h>

// ---------------- constants ----------------
#define T_TOK 2048
#define DIM   2048
#define FFN   4096
#define NEXP  8
#define BM    128
#define BN    64
#define BK    32
#define TP    3072            // T + NEXP*BM padded rows
#define MAXTILES 24           // T/BM + NEXP

typedef __bf16 v16bf __attribute__((ext_vector_type(16)));
typedef __bf16 v8bf  __attribute__((ext_vector_type(8)));
typedef float  v8f   __attribute__((ext_vector_type(8)));

__device__ __forceinline__ v16bf mk16(v8bf a, v8bf b) {
  v16bf r;
#pragma unroll
  for (int i = 0; i < 8; ++i) { r[i] = a[i]; r[i + 8] = b[i]; }
  return r;
}

// A fragment (16x32 bf16): lane l holds row M=l%16; K chunks [h*8,h*8+8) and [16+h*8,...)
__device__ __forceinline__ v16bf ldsA(const __bf16* s, int m, int lane) {
  const __bf16* r = s + m * BK;
  int h = (lane >> 4) & 1;
  v8bf c0 = *(const v8bf*)(r + h * 8);
  v8bf c1 = *(const v8bf*)(r + 16 + h * 8);
  return mk16(c0, c1);
}

// B fragment (32x16 bf16): lane l holds col N=l%16; K chunks [h*8,h*8+8) and [16+h*8,...)
// sB stored transposed: [BN][BK]
__device__ __forceinline__ v16bf ldsB(const __bf16* s, int n, int lane) {
  const __bf16* r = s + n * BK + ((lane >> 4) & 1) * 8;
  v8bf c0 = *(const v8bf*)(r);
  v8bf c1 = *(const v8bf*)(r + 16);
  return mk16(c0, c1);
}

// ---------------- init: zero workspace state ----------------
__global__ void init_kernel(unsigned int* xp32, int* perm, float* gwp,
                            int* counts, int* nTiles) {
  long gid = (long)blockIdx.x * blockDim.x + threadIdx.x;
  long stride = (long)gridDim.x * blockDim.x;
  const long XPW = (long)TP * DIM / 2;
  for (long i = gid; i < XPW; i += stride) xp32[i] = 0u;
  for (long i = gid; i < TP; i += stride) { perm[i] = -1; gwp[i] = 0.f; }
  if (gid < NEXP) counts[gid] = 0;
  if (gid == NEXP) *nTiles = 0;
}

// ---------------- router: logits -> top-1 -> sigmoid ----------------
__global__ __launch_bounds__(256) void router_kernel(
    const float* __restrict__ x, const float* __restrict__ rw,
    int* eid, float* gatew, int* rank, int* counts) {
  __shared__ float sRW[DIM * NEXP];
  int tid = threadIdx.x;
  for (int i = tid; i < DIM * NEXP; i += 256) sRW[i] = rw[i];
  __syncthreads();
  int wave = tid >> 5, lane = tid & 31;
  int t = blockIdx.x * 8 + wave;
  float s[NEXP];
#pragma unroll
  for (int e = 0; e < NEXP; ++e) s[e] = 0.f;
  for (int d = lane; d < DIM; d += 32) {
    float xv = x[(size_t)t * DIM + d];
    const float* r = &sRW[d * NEXP];
#pragma unroll
    for (int e = 0; e < NEXP; ++e) s[e] += xv * r[e];
  }
#pragma unroll
  for (int e = 0; e < NEXP; ++e)
    for (int off = 16; off; off >>= 1) s[e] += __shfl_xor(s[e], off, 32);
  if (lane == 0) {
    int be = 0; float bv = s[0];
#pragma unroll
    for (int e = 1; e < NEXP; ++e) if (s[e] > bv) { bv = s[e]; be = e; }
    eid[t] = be;
    gatew[t] = 1.f / (1.f + __expf(-bv));
    rank[t] = atomicAdd(&counts[be], 1);
  }
}

// ---------------- plan: BM-aligned segments + tile table ----------------
__global__ void plan_kernel(const int* counts, int* astart,
                            int* tileE, int* tileM0, int* nTiles) {
  if (threadIdx.x == 0 && blockIdx.x == 0) {
    int p = 0, nt = 0;
    for (int e = 0; e < NEXP; ++e) {
      astart[e] = p;
      int c = counts[e];
      int k = (c + BM - 1) / BM;
      for (int i = 0; i < k; ++i) { tileE[nt] = e; tileM0[nt] = p + i * BM; ++nt; }
      p += k * BM;
    }
    *nTiles = nt;
  }
}

// ---------------- scatter: permute + fp32->bf16 convert ----------------
__global__ __launch_bounds__(256) void scatter_kernel(
    const float* __restrict__ x, const int* eid, const int* rank,
    const int* astart, const float* gatew,
    __bf16* xp, __bf16* xbf, int* perm, float* gwp) {
  int t = blockIdx.x;
  int e = eid[t];
  int p = astart[e] + rank[t];
  for (int j = threadIdx.x; j < DIM; j += 256) {
    __bf16 v = (__bf16)x[(size_t)t * DIM + j];
    xp[(size_t)p * DIM + j] = v;
    xbf[(size_t)t * DIM + j] = v;
  }
  if (threadIdx.x == 0) { perm[p] = t; gwp[p] = gatew[t]; }
}

// ---------------- GEMM1: H = silu(A@Wg) * (A@Wu), bf16 out ----------------
__global__ __launch_bounds__(256) void gemm1_kernel(
    const __bf16* __restrict__ A, const float* __restrict__ Wg,
    const float* __restrict__ Wu, __bf16* __restrict__ H,
    const int* __restrict__ tileE, const int* __restrict__ tileM0,
    const int* __restrict__ nTiles, long wStride) {
  __shared__ __bf16 sA[2][BM * BK];
  __shared__ __bf16 sBg[2][BN * BK];
  __shared__ __bf16 sBu[2][BN * BK];
  int tid = threadIdx.x, bx = blockIdx.x;
  int e = 0, m0;
  if (tileE) {
    if (bx >= *nTiles) return;
    e = tileE[bx]; m0 = tileM0[bx];
  } else m0 = bx * BM;
  const float* wg = Wg + (long)e * wStride;
  const float* wu = Wu + (long)e * wStride;
  int n0 = blockIdx.y * BN;
  int lane = tid & 31, wave = tid >> 5;
  int wm = wave & 3, wn = wave >> 2;

  v8f accg[2][2], accu[2][2];
#pragma unroll
  for (int i = 0; i < 2; ++i)
#pragma unroll
    for (int j = 0; j < 2; ++j)
#pragma unroll
      for (int r = 0; r < 8; ++r) { accg[i][j][r] = 0.f; accu[i][j][r] = 0.f; }

  uint4 raA[2]; float4 raG[2], raU[2];
  auto loadG = [&](int ks) {
#pragma unroll
    for (int j = 0; j < 2; ++j) {
      int c = tid + j * 256;
      int row = c >> 2, q = c & 3;
      raA[j] = *(const uint4*)(A + (size_t)(m0 + row) * DIM + ks * BK + q * 8);
      int kr = c >> 4, f4 = c & 15;
      raG[j] = *(const float4*)(wg + (size_t)(ks * BK + kr) * FFN + n0 + f4 * 4);
      raU[j] = *(const float4*)(wu + (size_t)(ks * BK + kr) * FFN + n0 + f4 * 4);
    }
  };
  auto storeL = [&](int buf) {
#pragma unroll
    for (int j = 0; j < 2; ++j) {
      int c = tid + j * 256;
      int row = c >> 2, q = c & 3;
      *((uint4*)(&sA[buf][row * BK]) + q) = raA[j];
      int kr = c >> 4, f4 = c & 15;
      __bf16* bg = &sBg[buf][(f4 * 4) * BK + kr];
      bg[0 * BK] = (__bf16)raG[j].x; bg[1 * BK] = (__bf16)raG[j].y;
      bg[2 * BK] = (__bf16)raG[j].z; bg[3 * BK] = (__bf16)raG[j].w;
      __bf16* bu = &sBu[buf][(f4 * 4) * BK + kr];
      bu[0 * BK] = (__bf16)raU[j].x; bu[1 * BK] = (__bf16)raU[j].y;
      bu[2 * BK] = (__bf16)raU[j].z; bu[3 * BK] = (__bf16)raU[j].w;
    }
  };

  loadG(0); storeL(0); __syncthreads();
  const int NK = DIM / BK;
  for (int ks = 0; ks < NK; ++ks) {
    int buf = ks & 1;
    if (ks + 1 < NK) loadG(ks + 1);
    if (ks + 2 < NK) __builtin_prefetch(wg + (size_t)((ks + 2) * BK) * FFN + n0, 0, 0);
    v16bf af[2], bgf[2], buf2[2];
#pragma unroll
    for (int s = 0; s < 2; ++s) {
      af[s]   = ldsA(&sA[buf][0],  wm * 32 + s * 16 + (lane & 15), lane);
      bgf[s]  = ldsB(&sBg[buf][0], wn * 32 + s * 16 + (lane & 15), lane);
      buf2[s] = ldsB(&sBu[buf][0], wn * 32 + s * 16 + (lane & 15), lane);
    }
#pragma unroll
    for (int i = 0; i < 2; ++i)
#pragma unroll
      for (int j = 0; j < 2; ++j) {
        accg[i][j] = __builtin_amdgcn_wmma_f32_16x16x32_bf16(
            false, af[i], false, bgf[j], (short)0, accg[i][j], false, false);
        accu[i][j] = __builtin_amdgcn_wmma_f32_16x16x32_bf16(
            false, af[i], false, buf2[j], (short)0, accu[i][j], false, false);
      }
    if (ks + 1 < NK) storeL(buf ^ 1);
    __syncthreads();
  }
  int lm = ((lane >> 4) & 1) * 8, ln = lane & 15;
#pragma unroll
  for (int i = 0; i < 2; ++i)
#pragma unroll
    for (int j = 0; j < 2; ++j)
#pragma unroll
      for (int r = 0; r < 8; ++r) {
        int row = m0 + wm * 32 + i * 16 + lm + r;
        int col = n0 + wn * 32 + j * 16 + ln;
        float g = accg[i][j][r], u = accu[i][j][r];
        float h = (g / (1.f + __expf(-g))) * u;
        H[(size_t)row * FFN + col] = (__bf16)h;
      }
}

// ---------------- GEMM2: Y = H @ Wd ; store (shared) or perm-scaled add ----------------
__global__ __launch_bounds__(256) void gemm2_kernel(
    const __bf16* __restrict__ A, const float* __restrict__ Wd,
    float* __restrict__ out,
    const int* __restrict__ tileE, const int* __restrict__ tileM0,
    const int* __restrict__ nTiles,
    const int* __restrict__ perm, const float* __restrict__ gwp, long wStride) {
  __shared__ __bf16 sA[2][BM * BK];
  __shared__ __bf16 sB[2][BN * BK];
  int tid = threadIdx.x, bx = blockIdx.x;
  int e = 0, m0;
  if (tileE) {
    if (bx >= *nTiles) return;
    e = tileE[bx]; m0 = tileM0[bx];
  } else m0 = bx * BM;
  const float* wd = Wd + (long)e * wStride;
  int n0 = blockIdx.y * BN;
  int lane = tid & 31, wave = tid >> 5;
  int wm = wave & 3, wn = wave >> 2;

  v8f acc[2][2];
#pragma unroll
  for (int i = 0; i < 2; ++i)
#pragma unroll
    for (int j = 0; j < 2; ++j)
#pragma unroll
      for (int r = 0; r < 8; ++r) acc[i][j][r] = 0.f;

  uint4 raA[2]; float4 raB[2];
  auto loadG = [&](int ks) {
#pragma unroll
    for (int j = 0; j < 2; ++j) {
      int c = tid + j * 256;
      int row = c >> 2, q = c & 3;
      raA[j] = *(const uint4*)(A + (size_t)(m0 + row) * FFN + ks * BK + q * 8);
      int kr = c >> 4, f4 = c & 15;
      raB[j] = *(const float4*)(wd + (size_t)(ks * BK + kr) * DIM + n0 + f4 * 4);
    }
  };
  auto storeL = [&](int buf) {
#pragma unroll
    for (int j = 0; j < 2; ++j) {
      int c = tid + j * 256;
      int row = c >> 2, q = c & 3;
      *((uint4*)(&sA[buf][row * BK]) + q) = raA[j];
      int kr = c >> 4, f4 = c & 15;
      __bf16* bb = &sB[buf][(f4 * 4) * BK + kr];
      bb[0 * BK] = (__bf16)raB[j].x; bb[1 * BK] = (__bf16)raB[j].y;
      bb[2 * BK] = (__bf16)raB[j].z; bb[3 * BK] = (__bf16)raB[j].w;
    }
  };

  loadG(0); storeL(0); __syncthreads();
  const int NK = FFN / BK;
  for (int ks = 0; ks < NK; ++ks) {
    int buf = ks & 1;
    if (ks + 1 < NK) loadG(ks + 1);
    if (ks + 2 < NK) __builtin_prefetch(wd + (size_t)((ks + 2) * BK) * DIM + n0, 0, 0);
    v16bf af[2], bf2[2];
#pragma unroll
    for (int s = 0; s < 2; ++s) {
      af[s]  = ldsA(&sA[buf][0], wm * 32 + s * 16 + (lane & 15), lane);
      bf2[s] = ldsB(&sB[buf][0], wn * 32 + s * 16 + (lane & 15), lane);
    }
#pragma unroll
    for (int i = 0; i < 2; ++i)
#pragma unroll
      for (int j = 0; j < 2; ++j)
        acc[i][j] = __builtin_amdgcn_wmma_f32_16x16x32_bf16(
            false, af[i], false, bf2[j], (short)0, acc[i][j], false, false);
    if (ks + 1 < NK) storeL(buf ^ 1);
    __syncthreads();
  }
  int lm = ((lane >> 4) & 1) * 8, ln = lane & 15;
#pragma unroll
  for (int i = 0; i < 2; ++i)
#pragma unroll
    for (int j = 0; j < 2; ++j)
#pragma unroll
      for (int r = 0; r < 8; ++r) {
        int row = m0 + wm * 32 + i * 16 + lm + r;
        int col = n0 + wn * 32 + j * 16 + ln;
        float v = acc[i][j][r];
        if (perm) {
          int t = perm[row];
          if (t >= 0) {
            float* o = out + (size_t)t * DIM + col;
            *o = *o + gwp[row] * v;
          }
        } else {
          out[(size_t)row * DIM + col] = v;
        }
      }
}

// ---------------- launch ----------------
extern "C" void kernel_launch(void* const* d_in, const int* in_sizes, int n_in,
                              void* d_out, int out_size, void* d_ws, size_t ws_size,
                              hipStream_t stream) {
  const float* x       = (const float*)d_in[0];
  const float* rw      = (const float*)d_in[1];
  const float* w_gate  = (const float*)d_in[2];
  const float* w_up    = (const float*)d_in[3];
  const float* w_down  = (const float*)d_in[4];
  const float* ws_gate = (const float*)d_in[5];
  const float* ws_up   = (const float*)d_in[6];
  const float* ws_down = (const float*)d_in[7];
  float* out = (float*)d_out;

  char* ws = (char*)d_ws;
  auto carve = [&](size_t bytes) -> void* {
    void* p = (void*)ws;
    ws += (bytes + 255) & ~(size_t)255;
    return p;
  };
  __bf16* xp   = (__bf16*)carve((size_t)TP * DIM * 2);
  __bf16* xbf  = (__bf16*)carve((size_t)T_TOK * DIM * 2);
  __bf16* hp   = (__bf16*)carve((size_t)TP * FFN * 2);
  __bf16* hs   = (__bf16*)carve((size_t)T_TOK * FFN * 2);
  int*    perm = (int*)carve(TP * 4);
  float*  gwp  = (float*)carve(TP * 4);
  int*    eid  = (int*)carve(T_TOK * 4);
  int*    rank = (int*)carve(T_TOK * 4);
  float*  gatw = (float*)carve(T_TOK * 4);
  int*    counts = (int*)carve(16 * 4);
  int*    astart = (int*)carve(16 * 4);
  int*    tileE  = (int*)carve(32 * 4);
  int*    tileM0 = (int*)carve(32 * 4);
  int*    nTiles = (int*)carve(4);

  init_kernel<<<2048, 256, 0, stream>>>((unsigned int*)xp, perm, gwp, counts, nTiles);
  router_kernel<<<T_TOK / 8, 256, 0, stream>>>(x, rw, eid, gatw, rank, counts);
  plan_kernel<<<1, 32, 0, stream>>>(counts, astart, tileE, tileM0, nTiles);
  scatter_kernel<<<T_TOK, 256, 0, stream>>>(x, eid, rank, astart, gatw, xp, xbf, perm, gwp);

  // shared expert: gate/up -> hs
  gemm1_kernel<<<dim3(T_TOK / BM, FFN / BN), 256, 0, stream>>>(
      xbf, ws_gate, ws_up, hs, nullptr, nullptr, nullptr, 0L);
  // routed experts: gate/up -> hp
  gemm1_kernel<<<dim3(MAXTILES, FFN / BN), 256, 0, stream>>>(
      xp, w_gate, w_up, hp, tileE, tileM0, nTiles, (long)DIM * FFN);
  // shared down-proj: writes out
  gemm2_kernel<<<dim3(T_TOK / BM, DIM / BN), 256, 0, stream>>>(
      hs, ws_down, out, nullptr, nullptr, nullptr, nullptr, nullptr, 0L);
  // routed down-proj: out[t] += gw * y (stream-ordered after shared store)
  gemm2_kernel<<<dim3(MAXTILES, DIM / BN), 256, 0, stream>>>(
      hp, w_down, out, tileE, tileM0, nTiles, perm, gwp, (long)FFN * DIM);
}
